// fi_FFTConv_52776558133825
// MI455X (gfx1250) — matrix operand order repaired and checked by reading, not compiled, not run on hardware
//
#include <hip/hip_runtime.h>
#include <stdint.h>

typedef __attribute__((ext_vector_type(16))) __bf16 v16bf;
typedef __attribute__((ext_vector_type(8)))  float  v8f;

constexpr int kB   = 8;      // batch
constexpr int kT   = 2048;   // output length
constexpr int kCI  = 256;    // in channels
constexpr int kCO  = 256;    // out channels
constexpr int kK   = 65;     // kernel taps
constexpr int kLP  = 2112;   // padded signal length (2048 + 2*32)
constexpr int kTileCO = 128;
constexpr int kTileT  = 128;
constexpr int kSTL    = 192; // staged t-window: kTileT + 64
constexpr int kSROW   = 48;  // LDS signal row stride (elems): 96B rows, 32B aligned frags
constexpr int kWROW   = 40;  // LDS weight row stride (elems): 80B rows, 16B aligned frags

__device__ __forceinline__ unsigned short f2bf(float f) {
    unsigned u = __float_as_uint(f);
    u += 0x7FFFu + ((u >> 16) & 1u);      // round-to-nearest-even
    return (unsigned short)(u >> 16);
}

// Async bulk copy: 16 bytes global -> LDS, ASYNCcnt-tracked.
__device__ __forceinline__ void async_copy_b128(unsigned lds_byte_off,
                                                const unsigned short* gptr) {
    unsigned long long ga = (unsigned long long)(uintptr_t)gptr;
    asm volatile("global_load_async_to_lds_b128 %0, %1, off"
                 :: "v"(lds_byte_off), "v"(ga) : "memory");
}
__device__ __forceinline__ void async_wait0() {
    asm volatile("s_wait_asynccnt 0x0" ::: "memory");
}

// Pad + convert signal to bf16 in fragment-friendly order:
// sbf2[((b*8 + cib)*kLP + x)*32 + ci] = s_pad[b, cib*32+ci, x]
__global__ void prep_signal(const float* __restrict__ sig, unsigned short* __restrict__ sbf2) {
    size_t i = (size_t)blockIdx.x * 256 + threadIdx.x;
    const size_t total = (size_t)kB * kCI * kLP;
    if (i >= total) return;
    int ci = (int)(i & 31);
    size_t r = i >> 5;
    int x  = (int)(r % kLP);
    int bc = (int)(r / kLP);              // b*8 + cib
    int b = bc >> 3, cib = bc & 7;
    int ch = cib * 32 + ci;
    float v = 0.0f;
    if (x >= 32 && x < 32 + kT) v = sig[((size_t)b * kCI + ch) * kT + (x - 32)];
    sbf2[i] = f2bf(v);
}

// Transform weight w[co][ci][m] -> wt[m][ci>>5][co][ci&31] in bf16
__global__ void prep_weight(const float* __restrict__ w, unsigned short* __restrict__ wt) {
    size_t i = (size_t)blockIdx.x * 256 + threadIdx.x;
    const size_t total = (size_t)kCO * kCI * kK;
    if (i >= total) return;
    int m  = (int)(i % kK);
    size_t r = i / kK;
    int ci = (int)(r % kCI);
    int co = (int)(r / kCI);
    wt[(((size_t)m * 8 + (ci >> 5)) * kCO + co) * 32 + (ci & 31)] = f2bf(w[i]);
}

// out[b,co,t] = bias[co] + sum_ci sum_m s_pad[b,ci,t+m] * w[co,ci,m]
__global__ __launch_bounds__(256) void conv_wmma(
    const unsigned short* __restrict__ sbf,
    const unsigned short* __restrict__ wt,
    const float* __restrict__ bias,
    float* __restrict__ out)
{
    __shared__ unsigned short lds_sig[kSTL * kSROW];          // [t][ci], 18 KB
    __shared__ unsigned short lds_w[2][kTileCO * kWROW];      // [co][ci], 2 x 10 KB

    const int tid  = threadIdx.x;
    const int lane = tid & 31;
    const int lo = lane & 15;       // row/col within 16
    const int hi = lane >> 4;       // lane-half selects K/M half (ISA 7.12.2)
    const int wv  = tid >> 5;       // wave id 0..7
    const int wco = wv & 3;         // 4 waves along co (32 each)
    const int wtt = wv >> 2;        // 2 waves along t  (64 each)

    const int t0  = blockIdx.x * kTileT;
    const int co0 = blockIdx.y * kTileCO;
    const int b   = blockIdx.z;

    v8f acc[2][4];
    #pragma unroll
    for (int c = 0; c < 2; ++c)
        #pragma unroll
        for (int f = 0; f < 4; ++f)
            #pragma unroll
            for (int r = 0; r < 8; ++r) acc[c][f][r] = 0.0f;

    union AF { v16bf v; uint4 q[2]; };

    for (int cib = 0; cib < 8; ++cib) {
        __syncthreads();  // all waves done reading lds_sig / lds_w[0] of prev block

        // ---- async stage signal tile: 192 rows x 64B -> LDS rows of stride 96B ----
        {
            const unsigned short* gbase =
                sbf + (((size_t)b * 8 + cib) * kLP + t0) * 32;
            #pragma unroll
            for (int k = 0; k < 3; ++k) {
                int c = tid + k * 256;          // 0..767 = 192 rows * 4 chunks
                int row = c >> 2;               // t index 0..191
                int q   = c & 3;                // 16B chunk within row
                unsigned loff =
                    (unsigned)(uintptr_t)&lds_sig[row * kSROW + q * 8];
                async_copy_b128(loff, gbase + (size_t)row * 32 + q * 8);
            }
        }
        // ---- async stage weights m=0 into buffer 0: 128 rows x 64B ----
        {
            const unsigned short* gbase = wt + (((size_t)cib) * kCO + co0) * 32;
            #pragma unroll
            for (int k = 0; k < 2; ++k) {
                int c = tid + k * 256;          // 0..511 = 128 rows * 4 chunks
                int co = c >> 2, q = c & 3;
                unsigned loff =
                    (unsigned)(uintptr_t)&lds_w[0][co * kWROW + q * 8];
                async_copy_b128(loff, gbase + (size_t)co * 32 + q * 8);
            }
        }
        async_wait0();
        __syncthreads();

        for (int m = 0; m < kK; ++m) {
            const int cur = m & 1, nxt = cur ^ 1;
            const bool pf = (m + 1 < kK);
            if (pf) {   // async prefetch next-m weights into the other buffer
                const unsigned short* gbase =
                    wt + (((size_t)(m + 1) * 8 + cib) * kCO + co0) * 32;
                #pragma unroll
                for (int k = 0; k < 2; ++k) {
                    int c = tid + k * 256;
                    int co = c >> 2, q = c & 3;
                    unsigned loff =
                        (unsigned)(uintptr_t)&lds_w[nxt][co * kWROW + q * 8];
                    async_copy_b128(loff, gbase + (size_t)co * 32 + q * 8);
                }
            }

            // A fragments: weight rows (M=co), K=ci. lane lo = M, hi = K half.
            AF a0, a1;
            {
                const unsigned short* p0 = &lds_w[cur][(wco * 32 + lo) * kWROW + hi * 8];
                a0.q[0] = *(const uint4*)p0;          // K = hi*8 + 0..7
                a0.q[1] = *(const uint4*)(p0 + 16);   // K = 16 + hi*8 + 0..7
                const unsigned short* p1 = &lds_w[cur][(wco * 32 + 16 + lo) * kWROW + hi * 8];
                a1.q[0] = *(const uint4*)p1;
                a1.q[1] = *(const uint4*)(p1 + 16);
            }

            // B fragments: signal columns (N=t), K=ci. lane lo = N, hi = K half.
            #pragma unroll
            for (int f = 0; f < 4; ++f) {
                const unsigned short* pb =
                    &lds_sig[(wtt * 64 + f * 16 + lo + m) * kSROW + hi * 8];
                AF bu; bu.q[0] = *(const uint4*)pb; bu.q[1] = *(const uint4*)(pb + 16);   // K = hi*8 + 0..7, 16 + hi*8 + 0..7
                v16bf bv = bu.v;
                acc[0][f] = __builtin_amdgcn_wmma_f32_16x16x32_bf16(
                    false, a0.v, false, bv, (short)0, acc[0][f], false, false);
                acc[1][f] = __builtin_amdgcn_wmma_f32_16x16x32_bf16(
                    false, a1.v, false, bv, (short)0, acc[1][f], false, false);
            }

            if (pf) async_wait0();   // next-m weights landed (overlapped with WMMAs)
            __syncthreads();
        }
    }

    // ---- epilogue: bias + store f32. C/D layout: VGPR r -> M = hi*8 + r ----
    const int cobase = co0 + wco * 32 + hi * 8;
    #pragma unroll
    for (int c = 0; c < 2; ++c) {
        #pragma unroll
        for (int f = 0; f < 4; ++f) {
            const int tg = t0 + wtt * 64 + f * 16 + lo;
            #pragma unroll
            for (int r = 0; r < 8; ++r) {
                const int co = cobase + c * 16 + r;
                out[((size_t)b * kCO + co) * kT + tg] = acc[c][f][r] + bias[co];
            }
        }
    }
}

extern "C" void kernel_launch(void* const* d_in, const int* in_sizes, int n_in,
                              void* d_out, int out_size, void* d_ws, size_t ws_size,
                              hipStream_t stream) {
    const float* sig  = (const float*)d_in[0];   // (8,256,2048) f32
    const float* w    = (const float*)d_in[1];   // (256,256,65) f32
    const float* bias = (const float*)d_in[2];   // (256,) f32
    float* out = (float*)d_out;                  // (8,256,2048) f32

    unsigned short* sbf = (unsigned short*)d_ws;
    const size_t sbf_elems = (size_t)kB * kCI * kLP;        // 4,325,376 (8.65 MB)
    unsigned short* wtb = sbf + sbf_elems;                  // 4,259,840 (8.52 MB)

    {
        const size_t total = sbf_elems;
        prep_signal<<<dim3((unsigned)((total + 255) / 256)), dim3(256), 0, stream>>>(sig, sbf);
    }
    {
        const size_t total = (size_t)kCO * kCI * kK;
        prep_weight<<<dim3((unsigned)((total + 255) / 256)), dim3(256), 0, stream>>>(w, wtb);
    }
    dim3 grid(kT / kTileT, kCO / kTileCO, kB);   // (16, 2, 8)
    conv_wmma<<<grid, dim3(256), 0, stream>>>(sbf, wtb, bias, out);
}
